// Transformer_49873160241192
// MI455X (gfx1250) — matrix operand order repaired and checked by reading, not compiled, not run on hardware
//
#include <hip/hip_runtime.h>
#include <hip/hip_bf16.h>

// ---------------------------------------------------------------------------
// Transformer block (embed -> QKV proj -> reverse-causal attention -> ReLU)
// MI455X / gfx1250: bf16 WMMA, wave32, 8 waves/block, all fragment traffic
// as contiguous 2x b128 per lane (global + LDS).
// ---------------------------------------------------------------------------

#define BATCH 4
#define SEQ   4096
#define DIM   256

typedef __attribute__((ext_vector_type(16))) __bf16 v16bf;
typedef __attribute__((ext_vector_type(8)))  float  v8f;

union Frag16 { v16bf v; uint4 q[2]; };

// ---- WMMA wrapper ----------------------------------------------------------
__device__ __forceinline__ v8f wmma_bf16(v16bf a, v16bf b, v8f c) {
  return __builtin_amdgcn_wmma_f32_16x16x32_bf16(
      false, a, false, b, (short)0, c, false, false);
}

// ---- contiguous 16-bf16 fragment load (2 x b128) ---------------------------
__device__ __forceinline__ v16bf load16_contig(const __bf16* p) {
  Frag16 f;
  f.q[0] = *(const uint4*)(p);
  f.q[1] = *(const uint4*)(p + 8);
  return f.v;
}

// ---- A-matrix: 16x32 bf16 tile from row-major storage ----------------------
// Lane l<16: M=l, K=0..15 ; lane>=16: M=l-16, K=16..31
__device__ __forceinline__ v16bf load_A(const __bf16* base, int stride, int kOff, int lane) {
  int m  = lane & 15;
  int kb = (lane < 16) ? 0 : 16;
  const __bf16* row = base + m * stride + kOff + kb;
  Frag16 f;
  f.q[0] = *(const uint4*)(row);       // K = kb+0 .. kb+7
  f.q[1] = *(const uint4*)(row + 8);   // K = kb+8 .. kb+15
  return f.v;
}

// ---- B-matrix from "transposed" storage: element (k,n) = base[n*ld + k] ----
// Per lane: 16 contiguous bf16 along k -> 2 x b128.
__device__ __forceinline__ v16bf load_BT(const __bf16* base, long ld, int lane) {
  int n  = lane & 15;
  int kb = (lane < 16) ? 0 : 16;
  return load16_contig(base + (long)n * ld + kb);
}

// ===========================================================================
// Kernel 1: convert + swizzle Wq|Wk|Wv (f32) into B-fragment order:
//   wsW[(((mtx*8 + kt)*16 + nt)*32 + lane)*16 + j]
//   lane: n = lane&15, kb = (lane<16 ? 0 : 16);  j: K = kt*32 + kb + j
// ===========================================================================
__global__ __launch_bounds__(256) void pack_w_kernel(
    const float* __restrict__ Wq, const float* __restrict__ Wk,
    const float* __restrict__ Wv, __bf16* __restrict__ wsW) {
  int g = blockIdx.x * 256 + threadIdx.x;   // 0 .. 3*8*16*32-1
  const int NG = 3 * 8 * 16 * 32;
  if (g >= NG) return;
  int lane = g & 31;
  int nt   = (g >> 5) & 15;
  int kt   = (g >> 9) & 7;
  int mtx  = g >> 12;
  const float* W = (mtx == 0) ? Wq : ((mtx == 1) ? Wk : Wv);
  int n  = nt * 16 + (lane & 15);
  int kb = kt * 32 + ((lane < 16) ? 0 : 16);
  union { __bf16 h[16]; uint4 q[2]; } u;
#pragma unroll
  for (int j = 0; j < 16; ++j)
    u.h[j] = (__bf16)W[(long)(kb + j) * DIM + n];
  uint4* dst = (uint4*)wsW + (long)g * 2;
  dst[0] = u.q[0];
  dst[1] = u.q[1];
}

// ===========================================================================
// Kernel 2: fused embedding-gather + QKV projection (bf16 WMMA, f32 acc)
// Q,K stored row-major [B][S][D]; V stored transposed [B][D][S].
// ===========================================================================
__global__ __launch_bounds__(256) void qkv_kernel(
    const int* __restrict__ x, const float* __restrict__ emb,
    const __bf16* __restrict__ wsW,
    const float* __restrict__ bq, const float* __restrict__ bk,
    const float* __restrict__ bv,
    __bf16* __restrict__ Qbf, __bf16* __restrict__ Kbf, __bf16* __restrict__ Vt) {
  __shared__ __bf16 hA[16 * DIM];      // 8 KB activations (bf16)
  __shared__ __bf16 sStage[8][256];    // 4 KB per-wave C-tile staging
  __shared__ int tok[16];

  int tid = threadIdx.x;
  int bb  = blockIdx.x / (SEQ / 16);
  int s0  = (blockIdx.x % (SEQ / 16)) * 16;

  if (tid < 16) tok[tid] = x[bb * SEQ + s0 + tid];
  __syncthreads();

  // Gather 16 embedding rows (float4 loads), convert, pack (uint2 LDS stores)
  for (int i4 = tid; i4 < 16 * DIM / 4; i4 += 256) {
    int r  = i4 >> 6;            // 64 float4 per row
    int c4 = (i4 & 63) * 4;
    float4 f = *(const float4*)(emb + (long)tok[r] * DIM + c4);
    union { __bf16 h[4]; uint2 u2; } u;
    u.h[0] = (__bf16)f.x; u.h[1] = (__bf16)f.y;
    u.h[2] = (__bf16)f.z; u.h[3] = (__bf16)f.w;
    *(uint2*)(hA + r * DIM + c4) = u.u2;
  }
  __syncthreads();

  int lane = tid & 31, wave = tid >> 5;
  int hi = lane >> 4, n_lo = lane & 15;

  for (int t = wave; t < 48; t += 8) {       // uniform: 6 tiles per wave
    int mtx = t >> 4;                        // 0=Q, 1=K, 2=V
    int nt  = t & 15;
    int n0  = nt << 4;
    const float* bias = (mtx == 0) ? bq : ((mtx == 1) ? bk : bv);

    v8f c;
#pragma unroll
    for (int r = 0; r < 8; ++r) c[r] = 0.0f;

#pragma unroll
    for (int kk = 0; kk < 8; ++kk) {         // K = 256 in steps of 32
      v16bf a = load_A(hA, DIM, kk * 32, lane);
      v16bf b = load16_contig(wsW + ((long)(((mtx * 8 + kk) * 16 + nt) * 32 + lane)) * 16);
      c = wmma_bf16(a, b, c);
    }

    float bval = bias[n0 + n_lo];

    if (mtx == 2) {
      // V: transposed store -> 8 consecutive bf16 along S = one b128 per lane
      union { __bf16 h[8]; uint4 q; } u;
#pragma unroll
      for (int r = 0; r < 8; ++r) u.h[r] = (__bf16)(c[r] + bval);
      *(uint4*)(Vt + ((long)bb * DIM + n0 + n_lo) * SEQ + s0 + hi * 8) = u.q;
    } else {
      // Q/K: stage C tile in LDS, then one b128 row-segment store per lane
      __bf16* dst = (mtx == 0) ? Qbf : Kbf;
#pragma unroll
      for (int r = 0; r < 8; ++r)
        sStage[wave][(r + hi * 8) * 16 + n_lo] = (__bf16)(c[r] + bval);
      asm volatile("s_wait_dscnt 0x0" ::: "memory");
      int row  = lane >> 1;
      int half = lane & 1;
      uint4 vq = *(const uint4*)(&sStage[wave][row * 16 + half * 8]);
      *(uint4*)(dst + ((long)bb * SEQ + s0 + row) * DIM + n0 + half * 8) = vq;
    }
  }
}

// ===========================================================================
// Kernel 3: flash attention, reverse-causal mask (keep t >= s), + ReLU
// 16 queries/block; 128 keys/iter (one score tile per wave);
// each wave owns output dims [wave*32, wave*32+32).
// ===========================================================================
__global__ __launch_bounds__(256) void attn_kernel(
    const __bf16* __restrict__ Qbf, const __bf16* __restrict__ Kbf,
    const __bf16* __restrict__ Vt, float* __restrict__ out) {
  __shared__ __align__(16) unsigned char smem[20480];
  __bf16* sQ = (__bf16*)smem;            // [0,8K)   query tile
  float*  sS = (float*)(smem + 8192);    // [8K,16K) raw scores
  __bf16* sP = (__bf16*)(smem + 16384);  // [16K,20K) probabilities
  __shared__ float sM[16], sL[16], sAl[16];

  int tid = threadIdx.x, lane = tid & 31, wave = tid >> 5;
  int bb = blockIdx.x / (SEQ / 16);
  int s0 = (blockIdx.x % (SEQ / 16)) * 16;
  int hi = lane >> 4, n_lo = lane & 15;
  int d0 = wave * 32;
  const float scale = 0.0625f;  // 1/sqrt(256)

  // Q tile is one contiguous 8KB region: copy with b128s
  {
    const uint4* src = (const uint4*)(Qbf + ((long)bb * SEQ + s0) * DIM);
    uint4* dst = (uint4*)sQ;
    for (int i = tid; i < 16 * DIM / 8; i += 256) dst[i] = src[i];
  }
  if (tid < 16) { sM[tid] = -1.0e30f; sL[tid] = 0.0f; sAl[tid] = 1.0f; }
  __syncthreads();

  // Hoist the 8 Q A-fragments into registers (64 VGPRs, loop-invariant)
  v16bf qf[8];
#pragma unroll
  for (int kk = 0; kk < 8; ++kk) qf[kk] = load_A(sQ, DIM, kk * 32, lane);

  v8f acc0, acc1;
#pragma unroll
  for (int r = 0; r < 8; ++r) { acc0[r] = 0.0f; acc1[r] = 0.0f; }

  // Aligned start: every row's diagonal key lands in the first chunk;
  // SEQ % 128 == 0 so all chunks are fully in-bounds.
  for (int t0 = s0 & ~127; t0 < SEQ; t0 += 128) {
    // ---- scores: S[16 x 128] = Q @ K^T, one 16x16 tile per wave ----
    int tk0 = t0 + wave * 16;
    const __bf16* Kp = Kbf + ((long)bb * SEQ + tk0) * DIM;
    if (t0 + 128 < SEQ)  // global_prefetch_b8 on next chunk's K rows
      __builtin_prefetch(Kbf + ((long)bb * SEQ + tk0 + 128) * DIM, 0, 1);

    v8f c;
#pragma unroll
    for (int r = 0; r < 8; ++r) c[r] = 0.0f;
#pragma unroll
    for (int kk = 0; kk < 8; ++kk) {
      v16bf b = load_BT(Kp + kk * 32, DIM, lane);  // contiguous along dim
      c = wmma_bf16(qf[kk], b, c);
    }
    int tcol = tk0 + n_lo;
#pragma unroll
    for (int r = 0; r < 8; ++r) {
      int M = r + hi * 8;
      float v = (tcol >= s0 + M) ? c[r] * scale : -1.0e30f;
      sS[M * 128 + wave * 16 + n_lo] = v;
    }
    __syncthreads();

    // ---- online softmax stats: wave w handles rows 2w, 2w+1 ----
#pragma unroll
    for (int rr = 0; rr < 2; ++rr) {
      int row = wave * 2 + rr;
      float mloc = -1.0e30f;
#pragma unroll
      for (int j = 0; j < 4; ++j)
        mloc = fmaxf(mloc, sS[row * 128 + lane + 32 * j]);
#pragma unroll
      for (int off = 16; off > 0; off >>= 1)
        mloc = fmaxf(mloc, __shfl_xor(mloc, off, 32));

      float mold = sM[row];
      float mnew = fmaxf(mold, mloc);
      float ssum = 0.0f;
#pragma unroll
      for (int j = 0; j < 4; ++j) {
        float p = __expf(sS[row * 128 + lane + 32 * j] - mnew);
        ssum += p;
        sP[row * 128 + lane + 32 * j] = (__bf16)p;
      }
#pragma unroll
      for (int off = 16; off > 0; off >>= 1)
        ssum += __shfl_xor(ssum, off, 32);

      if (lane == 0) {
        float al = __expf(mold - mnew);
        sM[row]  = mnew;
        sL[row]  = sL[row] * al + ssum;
        sAl[row] = al;
      }
    }
    __syncthreads();

    // ---- rescale accumulators, then O += P @ V ----
#pragma unroll
    for (int r = 0; r < 8; ++r) {
      float al = sAl[r + hi * 8];
      acc0[r] *= al;
      acc1[r] *= al;
    }
    // V is stored transposed [B][D][S]: B-fragments contiguous along keys
    const __bf16* Vb0 = Vt + ((long)bb * DIM + d0) * SEQ + t0;
    const __bf16* Vb1 = Vt + ((long)bb * DIM + d0 + 16) * SEQ + t0;
#pragma unroll
    for (int kc = 0; kc < 4; ++kc) {
      v16bf a  = load_A(sP, 128, kc * 32, lane);
      v16bf b0 = load_BT(Vb0 + kc * 32, SEQ, lane);
      acc0 = wmma_bf16(a, b0, acc0);
      v16bf b1 = load_BT(Vb1 + kc * 32, SEQ, lane);
      acc1 = wmma_bf16(a, b1, acc1);
    }
    __syncthreads();
  }

  // ---- epilogue: normalize, ReLU, stage f32 tile in LDS, b128 stores ----
  float* sOut = (float*)smem;  // 16 KB, reuses sQ+sS after final barrier
#pragma unroll
  for (int r = 0; r < 8; ++r) {
    int M = r + hi * 8;
    float inv = 1.0f / sL[M];
    sOut[M * DIM + d0 + n_lo]      = fmaxf(acc0[r] * inv, 0.0f);
    sOut[M * DIM + d0 + 16 + n_lo] = fmaxf(acc1[r] * inv, 0.0f);
  }
  __syncthreads();
  {
    uint4* dst = (uint4*)(out + ((long)bb * SEQ + s0) * DIM);  // contiguous 16KB
    const uint4* src = (const uint4*)sOut;
    for (int i = tid; i < 16 * DIM / 4; i += 256) dst[i] = src[i];
  }
}

// ===========================================================================
// Host-side launch
// ===========================================================================
extern "C" void kernel_launch(void* const* d_in, const int* in_sizes, int n_in,
                              void* d_out, int out_size, void* d_ws, size_t ws_size,
                              hipStream_t stream) {
  const int*   x   = (const int*)d_in[0];
  const float* emb = (const float*)d_in[1];
  const float* Wq  = (const float*)d_in[2];
  const float* bq  = (const float*)d_in[3];
  const float* Wk  = (const float*)d_in[4];
  const float* bk  = (const float*)d_in[5];
  const float* Wv  = (const float*)d_in[6];
  const float* bv  = (const float*)d_in[7];
  float* out = (float*)d_out;

  // Workspace (bf16): [ Wpacked 3*D*D | Q B*S*D | K B*S*D | V^T B*D*S ]
  char* ws = (char*)d_ws;
  __bf16* wsW = (__bf16*)ws;
  size_t wBytes   = (size_t)3 * DIM * DIM * sizeof(__bf16);      // 384 KB
  size_t qkvBytes = (size_t)BATCH * SEQ * DIM * sizeof(__bf16);  // 8 MB each
  __bf16* Qbf = (__bf16*)(ws + wBytes);
  __bf16* Kbf = (__bf16*)(ws + wBytes + qkvBytes);
  __bf16* Vt  = (__bf16*)(ws + wBytes + 2 * qkvBytes);

  pack_w_kernel<<<(3 * 8 * 16 * 32 + 255) / 256, 256, 0, stream>>>(Wq, Wk, Wv, wsW);

  int nTiles = BATCH * (SEQ / 16);  // 1024 blocks
  qkv_kernel<<<nTiles, 256, 0, stream>>>(x, emb, wsW, bq, bk, bv, Qbf, Kbf, Vt);
  attn_kernel<<<nTiles, 256, 0, stream>>>(Qbf, Kbf, Vt, out);
}
